// OLGA_33483565040018
// MI455X (gfx1250) — matrix operand order repaired and checked by reading, not compiled, not run on hardware
//
#include <hip/hip_runtime.h>
#include <hip/hip_bf16.h>
#include <math.h>

// ---------------------------------------------------------------------------
// GCN forward for MI455X (gfx1250, wave32, WMMA).
//   h1 = tanh(gcnconv(x,  W1, b1));  h2 = tanh(gcnconv(h1, W2, b2))
// GEMMs run on v_wmma_f32_16x16x32_bf16 (bf16 in, f32 accumulate).
// Edge aggregation = float4 gathers + global f32 atomics (L2-resident).
// ---------------------------------------------------------------------------

typedef __attribute__((ext_vector_type(16))) __bf16 v16bf;
typedef __attribute__((ext_vector_type(8)))  float  v8f;

union FragBF { v16bf v; uint4 q[2]; };

__device__ __forceinline__ unsigned short f2bf(float f) {
  unsigned int u = __float_as_uint(f);
  unsigned int r = (u + 0x7FFFu + ((u >> 16) & 1u)) >> 16;  // RNE
  return (unsigned short)r;
}

// ---------------- degree / normalization ----------------
__global__ void k_deg_init(int* cnt, int n) {
  int i = blockIdx.x * blockDim.x + threadIdx.x;
  if (i < n) cnt[i] = 1;  // self loop
}

__global__ void k_deg_count(const long long* __restrict__ ei, int* cnt, int E) {
  int e = blockIdx.x * blockDim.x + threadIdx.x;
  if (e < E) atomicAdd(&cnt[(int)ei[E + e]], 1);  // targets = row 1
}

__global__ void k_dinv(void* buf, int n) {
  int i = blockIdx.x * blockDim.x + threadIdx.x;
  if (i < n) {
    int c = ((const int*)buf)[i];
    ((float*)buf)[i] = rsqrtf((float)c);  // deg >= 1 always
  }
}

// ---------------- conversions ----------------
__global__ void k_f32_to_bf16(const float* __restrict__ src,
                              unsigned short* __restrict__ dst, int n) {
  int i = blockIdx.x * blockDim.x + threadIdx.x;
  if (i < n) dst[i] = f2bf(src[i]);
}

// W[k*N + n] (f32, KxN row-major) -> Wt[n*K + k] (bf16, NxK row-major)
__global__ void k_transpose_to_bf16(const float* __restrict__ W,
                                    unsigned short* __restrict__ Wt,
                                    int K, int N) {
  int i = blockIdx.x * blockDim.x + threadIdx.x;
  if (i < K * N) {
    int k = i / N, n = i - k * N;
    Wt[(long)n * K + k] = f2bf(W[i]);
  }
}

// ---------------- WMMA GEMM: C[M,N] = A[M,K] * Bt[N,K]^T ----------------
// One wave computes a 16x64 strip (4 n-tiles, A fragment reused 4x).
__global__ void __launch_bounds__(256)
k_gemm_bf16_wmma(const unsigned short* __restrict__ A,
                 const unsigned short* __restrict__ Bt,
                 float* __restrict__ C, int M, int N, int K) {
  const int lane    = threadIdx.x & 31;
  const int waveId  = (blockIdx.x << 3) + (threadIdx.x >> 5);
  const int nGroups = N >> 6;                 // 64-col groups
  const int mTile   = waveId / nGroups;
  const int nGroup  = waveId - mTile * nGroups;
  if (mTile >= (M >> 4)) return;              // wave-uniform exit (EXEC stays all-1)

  const int ln   = lane & 15;                 // A-row / B-col within tile
  const int half = lane >> 4;

  const unsigned short* aBase = A + (long)((mTile << 4) + ln) * K;

  v8f acc0 = {}, acc1 = {}, acc2 = {}, acc3 = {};

  for (int k0 = 0; k0 < K; k0 += 32) {
    // A 16x32 bf16 fragment (ISA 7.12.2): elems 0-7 -> K = k0+half*8+e,
    //                                     elems 8-15 -> K = k0+16+half*8+(e-8)
    FragBF a;
    a.q[0] = *(const uint4*)(aBase + k0 + half * 8);
    a.q[1] = *(const uint4*)(aBase + k0 + 16 + half * 8);

#pragma unroll
    for (int t = 0; t < 4; ++t) {
      const int nTile = (nGroup << 2) + t;
      // B 32x16 bf16 fragment: same K order as A (k0+half*8+e, k0+16+half*8+(e-8))
      const unsigned short* bBase = Bt + (long)((nTile << 4) + ln) * K + k0 + half * 8;
      FragBF b;
      b.q[0] = *(const uint4*)(bBase);
      b.q[1] = *(const uint4*)(bBase + 16);

      if (t == 0) acc0 = __builtin_amdgcn_wmma_f32_16x16x32_bf16(false, a.v, false, b.v, (short)0, acc0, false, false);
      if (t == 1) acc1 = __builtin_amdgcn_wmma_f32_16x16x32_bf16(false, a.v, false, b.v, (short)0, acc1, false, false);
      if (t == 2) acc2 = __builtin_amdgcn_wmma_f32_16x16x32_bf16(false, a.v, false, b.v, (short)0, acc2, false, false);
      if (t == 3) acc3 = __builtin_amdgcn_wmma_f32_16x16x32_bf16(false, a.v, false, b.v, (short)0, acc3, false, false);
    }
  }

  // C/D layout: VGPR r -> row r + 8*half, col = lane&15
  const long rowBase = (long)(mTile << 4) + (half << 3);
#pragma unroll
  for (int r = 0; r < 8; ++r) {
    float* cRow = C + (rowBase + r) * N + (nGroup << 6) + ln;
    cRow[0]  = acc0[r];
    cRow[16] = acc1[r];
    cRow[32] = acc2[r];
    cRow[48] = acc3[r];
  }
}

// ---------------- aggregation ----------------
// self-loop term: out[i,:] = hpre[i,:] * dinv[i]^2
__global__ void k_agg_init(const float* __restrict__ hpre,
                           const float* __restrict__ dinv,
                           float* __restrict__ out, int n, int D) {
  int idx = blockIdx.x * blockDim.x + threadIdx.x;
  int chunks = D >> 2;
  if (idx >= n * chunks) return;
  int node = idx / chunks, j = idx - node * chunks;
  float s = dinv[node] * dinv[node];
  const float4 v = *(const float4*)(hpre + (long)node * D + (j << 2));
  float4 o; o.x = v.x * s; o.y = v.y * s; o.z = v.z * s; o.w = v.w * s;
  *(float4*)(out + (long)node * D + (j << 2)) = o;
}

// edge term: out[col,:] += hpre[row,:] * dinv[row]*dinv[col]
__global__ void k_agg_edges(const long long* __restrict__ ei,
                            const float* __restrict__ hpre,
                            const float* __restrict__ dinv,
                            float* __restrict__ out, int E, int D) {
  int idx = blockIdx.x * blockDim.x + threadIdx.x;
  int chunks = D >> 2;
  if (idx >= E * chunks) return;
  int e = idx / chunks, j = idx - e * chunks;
  int r = (int)ei[e];
  int c = (int)ei[E + e];
  float s = dinv[r] * dinv[c];
  const float4 v = *(const float4*)(hpre + (long)r * D + (j << 2));
  float* o = out + (long)c * D + (j << 2);
  atomicAdd(o + 0, v.x * s);
  atomicAdd(o + 1, v.y * s);
  atomicAdd(o + 2, v.z * s);
  atomicAdd(o + 3, v.w * s);
}

// ---------------- bias + tanh ----------------
__global__ void k_bias_tanh_bf16(const float* __restrict__ agg,
                                 const float* __restrict__ bias,
                                 unsigned short* __restrict__ outbf,
                                 int n, int D) {
  int i = blockIdx.x * blockDim.x + threadIdx.x;
  if (i < n * D) outbf[i] = f2bf(tanhf(agg[i] + bias[i % D]));
}

__global__ void k_bias_tanh_f32_inplace(float* __restrict__ agg,
                                        const float* __restrict__ bias,
                                        int n, int D) {
  int i = blockIdx.x * blockDim.x + threadIdx.x;
  if (i < n * D) agg[i] = tanhf(agg[i] + bias[i % D]);
}

// ---------------------------------------------------------------------------
extern "C" void kernel_launch(void* const* d_in, const int* in_sizes, int n_in,
                              void* d_out, int out_size, void* d_ws, size_t ws_size,
                              hipStream_t stream) {
  (void)in_sizes; (void)n_in; (void)out_size; (void)ws_size;

  const int N  = 50000;   // nodes (multiple of 16)
  const int E  = 800000;  // edges
  const int D0 = 256, D1 = 256, D2 = 128;

  const float*     x  = (const float*)d_in[0];
  const long long* ei = (const long long*)d_in[1];   // int64 [2, E]
  const float*     W1 = (const float*)d_in[2];
  const float*     b1 = (const float*)d_in[3];
  const float*     W2 = (const float*)d_in[4];
  const float*     b2 = (const float*)d_in[5];
  float*           out = (float*)d_out;              // [N, D2] f32

  // workspace layout
  char* ws = (char*)d_ws;
  size_t off = 0;
  auto wsAlloc = [&](size_t bytes) -> void* {
    off = (off + 255) & ~(size_t)255;
    void* p = ws + off;
    off += bytes;
    return p;
  };
  void*           degbuf = wsAlloc((size_t)N * 4);            // int cnt -> float dinv (in place)
  unsigned short* xbf    = (unsigned short*)wsAlloc((size_t)N * D0 * 2);
  unsigned short* w1t    = (unsigned short*)wsAlloc((size_t)D1 * D0 * 2);  // [N=D1][K=D0]
  unsigned short* w2t    = (unsigned short*)wsAlloc((size_t)D2 * D1 * 2);  // [N=D2][K=D1]
  float*          hpre   = (float*)wsAlloc((size_t)N * D1 * 4);            // reused layer 2
  float*          agg1   = (float*)wsAlloc((size_t)N * D1 * 4);
  unsigned short* h1bf   = (unsigned short*)wsAlloc((size_t)N * D1 * 2);

  const int T = 256;
  auto g = [](long n, int t) { return (int)((n + t - 1) / t); };

  // --- normalization coefficients ---
  k_deg_init <<<g(N, T), T, 0, stream>>>((int*)degbuf, N);
  k_deg_count<<<g(E, T), T, 0, stream>>>(ei, (int*)degbuf, E);
  k_dinv     <<<g(N, T), T, 0, stream>>>(degbuf, N);
  const float* dinv = (const float*)degbuf;

  // --- bf16 conversions / weight transpose ---
  k_f32_to_bf16     <<<g((long)N * D0, T), T, 0, stream>>>(x, xbf, N * D0);
  k_transpose_to_bf16<<<g((long)D0 * D1, T), T, 0, stream>>>(W1, w1t, D0, D1);
  k_transpose_to_bf16<<<g((long)D1 * D2, T), T, 0, stream>>>(W2, w2t, D1, D2);

  // --- layer 1 ---
  {
    int waves = (N / 16) * (D1 / 64);
    k_gemm_bf16_wmma<<<g(waves, 8), T, 0, stream>>>(xbf, w1t, hpre, N, D1, D0);
    k_agg_init <<<g((long)N * (D1 / 4), T), T, 0, stream>>>(hpre, dinv, agg1, N, D1);
    k_agg_edges<<<g((long)E * (D1 / 4), T), T, 0, stream>>>(ei, hpre, dinv, agg1, E, D1);
    k_bias_tanh_bf16<<<g((long)N * D1, T), T, 0, stream>>>(agg1, b1, h1bf, N, D1);
  }

  // --- layer 2 (aggregate straight into d_out, tanh in place) ---
  {
    int waves = (N / 16) * (D2 / 64);
    k_gemm_bf16_wmma<<<g(waves, 8), T, 0, stream>>>(h1bf, w2t, hpre, N, D2, D1);
    k_agg_init <<<g((long)N * (D2 / 4), T), T, 0, stream>>>(hpre, dinv, out, N, D2);
    k_agg_edges<<<g((long)E * (D2 / 4), T), T, 0, stream>>>(ei, hpre, dinv, out, E, D2);
    k_bias_tanh_f32_inplace<<<g((long)N * D2, T), T, 0, stream>>>(out, b2, N, D2);
  }
}
